// Seq2Seq_66082366816470
// MI455X (gfx1250) — compile-verified, matrix operand order checked by reading, not run on hardware
//
#include <hip/hip_runtime.h>
#include <math.h>

// ---------------------------------------------------------------------------
// Seq2Seq GRU (B=512, T_in=64, H=960, G=2880, T_pred=48) for gfx1250.
// All K=960 GEMMs run on v_wmma_f32_16x16x32_bf16. Weights are converted to
// bf16 once per launch; the recurrent state is maintained in BOTH f32 (for the
// elementwise GRU math) and bf16 (as GEMM A-operand), so the WMMA inner loop
// does pure b128 loads + WMMA with no conversion VALU.
// ---------------------------------------------------------------------------

typedef __attribute__((ext_vector_type(16))) __bf16 v16bf;
typedef __attribute__((ext_vector_type(8)))  __bf16 v8bf;
typedef __attribute__((ext_vector_type(8)))  float  v8f;

#define B_     512
#define TIN_   64
#define IN_    16
#define HS_    32
#define H_     960
#define G_     2880   // 3*H
#define TPRED_ 48
#define OUT_   32

static __device__ __forceinline__ __bf16 f2bf(float f) {
  union { float f; unsigned u; } v; v.f = f;
  unsigned u = v.u;
  unsigned r = (u + 0x7fffu + ((u >> 16) & 1u)) >> 16;   // round-to-nearest-even
  union { unsigned short s; __bf16 b; } o; o.s = (unsigned short)r;
  return o.b;
}

// --------------------------- WMMA GEMM -------------------------------------
// C[M,N] = A[M,K] (bf16, row-major) * W[N,K] (bf16, row-major)^T   (f32 accum)
// Block: 256 threads = 8 waves (4 M-waves x 2 N-waves).
// Block tile 64(M) x 64(N); each wave: one 16xK A-frag, two 16x16 C tiles.
// Requires: K % 32 == 0, M % 64 == 0, N % 32 == 0 (N may be < block tile).
__global__ __launch_bounds__(256)
void gemm_wmma_bf16(const __bf16* __restrict__ A,
                    const __bf16* __restrict__ W,
                    float* __restrict__ C,
                    int N, int K)
{
  const int wave = threadIdx.x >> 5;
  const int lane = threadIdx.x & 31;
  const int l    = lane & 15;
  const int hi   = lane >> 4;
  const int wm   = wave & 3;        // 4 wave rows (M)
  const int wn   = wave >> 2;       // 2 wave cols (each covers 32 N)
  const int tileM  = blockIdx.y * 64 + wm * 16;
  const int tileN0 = blockIdx.x * 64 + wn * 32;
  if (tileN0 >= N) return;          // uniform per wave (narrow-N case)
  const int tileN1 = tileN0 + 16;   // always valid: N % 32 == 0

  const __bf16* Arow  = A + (size_t)(tileM  + l) * (size_t)K;
  const __bf16* Wrow0 = W + (size_t)(tileN0 + l) * (size_t)K;
  const __bf16* Wrow1 = W + (size_t)(tileN1 + l) * (size_t)K;

  v8f acc0 = {}, acc1 = {};
#pragma unroll 2
  for (int k0 = 0; k0 < K; k0 += 32) {
    const int c0 = k0 + (hi << 3);  // lane-group K offset per ISA 16-bit layout
    v8bf a0  = *(const v8bf*)(Arow  + c0);
    v8bf a1  = *(const v8bf*)(Arow  + c0 + 16);
    v8bf b00 = *(const v8bf*)(Wrow0 + c0);
    v8bf b01 = *(const v8bf*)(Wrow0 + c0 + 16);
    v8bf b10 = *(const v8bf*)(Wrow1 + c0);
    v8bf b11 = *(const v8bf*)(Wrow1 + c0 + 16);
    v16bf a  = __builtin_shufflevector(a0,  a1,  0,1,2,3,4,5,6,7,8,9,10,11,12,13,14,15);
    v16bf b0 = __builtin_shufflevector(b00, b01, 0,1,2,3,4,5,6,7,8,9,10,11,12,13,14,15);
    v16bf b1 = __builtin_shufflevector(b10, b11, 0,1,2,3,4,5,6,7,8,9,10,11,12,13,14,15);
    acc0 = __builtin_amdgcn_wmma_f32_16x16x32_bf16(false, a, false, b0, (short)0, acc0, false, false);
    acc1 = __builtin_amdgcn_wmma_f32_16x16x32_bf16(false, a, false, b1, (short)0, acc1, false, false);
  }

  // C layout: VGPR r -> M = tileM + r + 8*hi, N = tileN + l
  float* Cr0 = C + (size_t)tileM * (size_t)N + (size_t)(tileN0 + l);
  float* Cr1 = C + (size_t)tileM * (size_t)N + (size_t)(tileN1 + l);
#pragma unroll
  for (int r = 0; r < 8; ++r) {
    Cr0[(size_t)(r + 8 * hi) * (size_t)N] = acc0[r];
    Cr1[(size_t)(r + 8 * hi) * (size_t)N] = acc1[r];
  }
}

// ------------------- small-K input projection (gi = x @ Wih^T) -------------
__global__ void gi_smallk_kernel(const float* __restrict__ X, int xstride_b, int xoff, int K,
                                 const float* __restrict__ Wih,   // G x K f32
                                 float* __restrict__ gi)
{
  int idx = blockIdx.x * blockDim.x + threadIdx.x;
  if (idx >= B_ * G_) return;
  int b = idx / G_, g = idx - b * G_;
  const float* x = X + (size_t)b * xstride_b + xoff;
  const float* w = Wih + (size_t)g * K;
  float s = 0.f;
  for (int k = 0; k < K; ++k) s += x[k] * w[k];
  gi[(size_t)b * G_ + g] = s;
}

// ------------------------------ GRU cell -----------------------------------
// Updates f32 state in place, mirrors it to bf16 (next GEMM's A operand),
// optionally scatters the new state into the hidden_outputs tensor.
__global__ void gru_cell_kernel(const float* __restrict__ gi,
                                const float* __restrict__ gh,
                                const float* __restrict__ bih,
                                const float* __restrict__ bhh,
                                float*  __restrict__ h,
                                __bf16* __restrict__ h_bf,
                                float*  __restrict__ hcat,     // optional (or null)
                                long hcat_stride_b)
{
  int idx = blockIdx.x * blockDim.x + threadIdx.x;
  if (idx >= B_ * H_) return;
  int b = idx / H_, j = idx - b * H_;
  size_t base = (size_t)b * G_;
  float ir = gi[base + j]          + bih[j];
  float iz = gi[base + H_ + j]     + bih[H_ + j];
  float in = gi[base + 2 * H_ + j] + bih[2 * H_ + j];
  float hr = gh[base + j]          + bhh[j];
  float hz = gh[base + H_ + j]     + bhh[H_ + j];
  float hn = gh[base + 2 * H_ + j] + bhh[2 * H_ + j];
  float r = 1.f / (1.f + expf(-(ir + hr)));
  float z = 1.f / (1.f + expf(-(iz + hz)));
  float n = tanhf(in + r * hn);
  float hv = h[(size_t)b * H_ + j];
  float out = (1.f - z) * n + z * hv;
  h[(size_t)b * H_ + j]    = out;
  h_bf[(size_t)b * H_ + j] = f2bf(out);
  if (hcat) hcat[(size_t)b * hcat_stride_b + j] = out;
}

// --------------------------- misc helpers ----------------------------------
__global__ void cvt_f32_bf16_kernel(const float* __restrict__ s, __bf16* __restrict__ d, int n) {
  int i = blockIdx.x * blockDim.x + threadIdx.x;
  if (i < n) d[i] = f2bf(s[i]);
}
__global__ void zero_f32_kernel(float* p, int n) {
  int i = blockIdx.x * blockDim.x + threadIdx.x;
  if (i < n) p[i] = 0.f;
}
__global__ void zero_bf16_kernel(unsigned short* p, int n) {
  int i = blockIdx.x * blockDim.x + threadIdx.x;
  if (i < n) p[i] = 0;
}
__global__ void copy_f32_kernel(const float* __restrict__ s, float* __restrict__ d, int n) {
  int i = blockIdx.x * blockDim.x + threadIdx.x;
  if (i < n) d[i] = s[i];
}
__global__ void out_bias_kernel(const float* __restrict__ tmp,
                                const float* __restrict__ lin_b,
                                float* __restrict__ outp, long ostride_b,
                                float* __restrict__ x_dec)
{
  int idx = blockIdx.x * blockDim.x + threadIdx.x;
  if (idx >= B_ * OUT_) return;
  int b = idx / OUT_, o = idx - b * OUT_;
  float v = tmp[(size_t)b * OUT_ + o] + lin_b[o];
  outp[(size_t)b * ostride_b + o] = v;
  x_dec[(size_t)b * OUT_ + o] = v;   // next decoder input
}

// ---------------------------------------------------------------------------
extern "C" void kernel_launch(void* const* d_in, const int* in_sizes, int n_in,
                              void* d_out, int out_size, void* d_ws, size_t ws_size,
                              hipStream_t stream) {
  (void)in_sizes; (void)n_in; (void)out_size; (void)ws_size;

  const float* in_data   = (const float*)d_in[0];
  const float* last_loc  = (const float*)d_in[1];
  //            d_in[2] = pred_length (known constant 48)
  const float* enc_Wih0  = (const float*)d_in[3];
  const float* enc_Whh0  = (const float*)d_in[4];
  const float* enc_bih0  = (const float*)d_in[5];
  const float* enc_bhh0  = (const float*)d_in[6];
  const float* enc_Wih1  = (const float*)d_in[7];
  const float* enc_Whh1  = (const float*)d_in[8];
  const float* enc_bih1  = (const float*)d_in[9];
  const float* enc_bhh1  = (const float*)d_in[10];
  const float* dec_Wih0  = (const float*)d_in[11];
  const float* dec_Whh0  = (const float*)d_in[12];
  const float* dec_bih0  = (const float*)d_in[13];
  const float* dec_bhh0  = (const float*)d_in[14];
  const float* dec_Wih1  = (const float*)d_in[15];
  const float* dec_Whh1  = (const float*)d_in[16];
  const float* dec_bih1  = (const float*)d_in[17];
  const float* dec_bhh1  = (const float*)d_in[18];
  const float* lin_W     = (const float*)d_in[19];
  const float* lin_b     = (const float*)d_in[20];

  // ----- workspace layout -----
  const size_t GK = (size_t)G_ * H_;        // 2,764,800
  char* ws = (char*)d_ws;
  size_t off = 0;
  auto take = [&](size_t bytes) { char* p = ws + off; off += (bytes + 255) & ~(size_t)255; return p; };

  __bf16* wb_enc_Whh0 = (__bf16*)take(GK * 2);
  __bf16* wb_enc_Wih1 = (__bf16*)take(GK * 2);
  __bf16* wb_enc_Whh1 = (__bf16*)take(GK * 2);
  __bf16* wb_dec_Whh0 = (__bf16*)take(GK * 2);
  __bf16* wb_dec_Wih1 = (__bf16*)take(GK * 2);
  __bf16* wb_dec_Whh1 = (__bf16*)take(GK * 2);
  __bf16* wb_lin      = (__bf16*)take((size_t)OUT_ * H_ * 2);

  float*  h0    = (float*) take((size_t)B_ * H_ * 4);
  float*  h1    = (float*) take((size_t)B_ * H_ * 4);
  __bf16* h0_bf = (__bf16*)take((size_t)B_ * H_ * 2);
  __bf16* h1_bf = (__bf16*)take((size_t)B_ * H_ * 2);
  float*  gi    = (float*) take((size_t)B_ * G_ * 4);
  float*  gh    = (float*) take((size_t)B_ * G_ * 4);
  float*  xdec  = (float*) take((size_t)B_ * HS_ * 4);
  float*  otmp  = (float*) take((size_t)B_ * OUT_ * 4);

  float* outs = (float*)d_out;                          // (B, 48, 32)
  float* hid  = outs + (size_t)B_ * TPRED_ * OUT_;      // (B, 48, 1920)

  const int T256 = 256;
  auto gridOf = [](size_t n) { return dim3((unsigned)((n + 255) / 256)); };

  // ----- one-time (per-launch) setup -----
  cvt_f32_bf16_kernel<<<gridOf(GK), T256, 0, stream>>>(enc_Whh0, wb_enc_Whh0, (int)GK);
  cvt_f32_bf16_kernel<<<gridOf(GK), T256, 0, stream>>>(enc_Wih1, wb_enc_Wih1, (int)GK);
  cvt_f32_bf16_kernel<<<gridOf(GK), T256, 0, stream>>>(enc_Whh1, wb_enc_Whh1, (int)GK);
  cvt_f32_bf16_kernel<<<gridOf(GK), T256, 0, stream>>>(dec_Whh0, wb_dec_Whh0, (int)GK);
  cvt_f32_bf16_kernel<<<gridOf(GK), T256, 0, stream>>>(dec_Wih1, wb_dec_Wih1, (int)GK);
  cvt_f32_bf16_kernel<<<gridOf(GK), T256, 0, stream>>>(dec_Whh1, wb_dec_Whh1, (int)GK);
  cvt_f32_bf16_kernel<<<gridOf((size_t)OUT_ * H_), T256, 0, stream>>>(lin_W, wb_lin, OUT_ * H_);

  zero_f32_kernel <<<gridOf((size_t)B_ * H_), T256, 0, stream>>>(h0, B_ * H_);
  zero_f32_kernel <<<gridOf((size_t)B_ * H_), T256, 0, stream>>>(h1, B_ * H_);
  zero_bf16_kernel<<<gridOf((size_t)B_ * H_), T256, 0, stream>>>((unsigned short*)h0_bf, B_ * H_);
  zero_bf16_kernel<<<gridOf((size_t)B_ * H_), T256, 0, stream>>>((unsigned short*)h1_bf, B_ * H_);
  copy_f32_kernel <<<gridOf((size_t)B_ * HS_), T256, 0, stream>>>(last_loc, xdec, B_ * HS_);

  const dim3 gemmGrid(G_ / 64, B_ / 64);     // (45, 8)
  const dim3 linGrid(1, B_ / 64);            // N=32 handled by wave early-out
  const dim3 cellGrid((unsigned)(((size_t)B_ * H_ + 255) / 256));
  const dim3 giGrid((unsigned)(((size_t)B_ * G_ + 255) / 256));

  // ----- encoder: 64 steps, 2 stacked GRU layers (interleaved) -----
  for (int t = 0; t < TIN_; ++t) {
    // layer 0: gi = x_t @ Wih0^T (K=16), gh = h0 @ Whh0^T (WMMA)
    gi_smallk_kernel<<<giGrid, T256, 0, stream>>>(in_data, TIN_ * IN_, t * IN_, IN_, enc_Wih0, gi);
    gemm_wmma_bf16<<<gemmGrid, T256, 0, stream>>>(h0_bf, wb_enc_Whh0, gh, G_, H_);
    gru_cell_kernel<<<cellGrid, T256, 0, stream>>>(gi, gh, enc_bih0, enc_bhh0, h0, h0_bf, nullptr, 0);
    // layer 1: x = new h0
    gemm_wmma_bf16<<<gemmGrid, T256, 0, stream>>>(h0_bf, wb_enc_Wih1, gi, G_, H_);
    gemm_wmma_bf16<<<gemmGrid, T256, 0, stream>>>(h1_bf, wb_enc_Whh1, gh, G_, H_);
    gru_cell_kernel<<<cellGrid, T256, 0, stream>>>(gi, gh, enc_bih1, enc_bhh1, h1, h1_bf, nullptr, 0);
  }

  // ----- decoder: 48 steps -----
  const long hidStrideB = (long)TPRED_ * 2 * H_;   // per-batch stride in hidden_outputs
  const long outStrideB = (long)TPRED_ * OUT_;
  for (int t = 0; t < TPRED_; ++t) {
    // layer 0 (input = 32-dim previous output)
    gi_smallk_kernel<<<giGrid, T256, 0, stream>>>(xdec, HS_, 0, HS_, dec_Wih0, gi);
    gemm_wmma_bf16<<<gemmGrid, T256, 0, stream>>>(h0_bf, wb_dec_Whh0, gh, G_, H_);
    gru_cell_kernel<<<cellGrid, T256, 0, stream>>>(gi, gh, dec_bih0, dec_bhh0, h0, h0_bf,
                                                   hid + (size_t)t * 2 * H_, hidStrideB);
    // layer 1
    gemm_wmma_bf16<<<gemmGrid, T256, 0, stream>>>(h0_bf, wb_dec_Wih1, gi, G_, H_);
    gemm_wmma_bf16<<<gemmGrid, T256, 0, stream>>>(h1_bf, wb_dec_Whh1, gh, G_, H_);
    gru_cell_kernel<<<cellGrid, T256, 0, stream>>>(gi, gh, dec_bih1, dec_bhh1, h1, h1_bf,
                                                   hid + (size_t)t * 2 * H_ + H_, hidStrideB);
    // linear head: out = h1 @ lin_W^T + b ; also feeds next step's input
    gemm_wmma_bf16<<<linGrid, T256, 0, stream>>>(h1_bf, wb_lin, otmp, OUT_, H_);
    out_bias_kernel<<<gridOf((size_t)B_ * OUT_), T256, 0, stream>>>(
        otmp, lin_b, outs + (size_t)t * OUT_, outStrideB, xdec);
  }
}
